// MultiHeadAttention_62766652064333
// MI455X (gfx1250) — matrix operand order repaired and checked by reading, not compiled, not run on hardware
//
#include <hip/hip_runtime.h>
#include <hip/hip_bf16.h>

typedef __attribute__((ext_vector_type(16))) __bf16 v16bf;
typedef __attribute__((ext_vector_type(8)))  __bf16 v8bf;
typedef __attribute__((ext_vector_type(8)))  float  v8f;

#if defined(__has_builtin)
#if __has_builtin(__builtin_amdgcn_global_load_async_to_lds_b128)
#define HAVE_ASYNC_LDS 1
#endif
#endif
#ifndef HAVE_ASYNC_LDS
#define HAVE_ASYNC_LDS 0
#endif

namespace {

constexpr int kDM = 512;
constexpr int kH  = 8;
constexpr int kB  = 8;
constexpr int kSQ = 1024;
constexpr int kSK = 1024;

constexpr int BM   = 128;      // block tile M
constexpr int BN   = 128;      // block tile N
constexpr int BK   = 32;       // K step (WMMA bf16 K)
constexpr int LDSP = BK + 8;   // padded LDS pitch (elements) -> conflict-free b128
constexpr int NT   = 256;      // 8 waves; wave grid 4(M) x 2(N), each 2x4 WMMA tiles

__device__ __forceinline__ unsigned short f32_to_bf16_bits(float f) {
  union { float f; unsigned u; } v; v.f = f;
  unsigned u = v.u;
  unsigned r = u + 0x7FFFu + ((u >> 16) & 1u);   // round-to-nearest-even
  return (unsigned short)(r >> 16);
}

#if HAVE_ASYNC_LDS
// Builtin signature (from clang diagnostic): first param is
// "__vector_size__(4*sizeof(int)) int" pointer in AS1 (__device__),
// second in AS3 (LDS), then imm offset, imm cpol.
typedef int v4i_gcc __attribute__((__vector_size__(16)));
typedef __attribute__((address_space(1))) v4i_gcc* g128_t;
typedef __attribute__((address_space(3))) v4i_gcc* l128_t;

__device__ __forceinline__ void async_b128(const unsigned short* g, unsigned short* l) {
  __builtin_amdgcn_global_load_async_to_lds_b128((g128_t)g, (l128_t)l, 0, 0);
}
__device__ __forceinline__ void wait_async0() {
#if __has_builtin(__builtin_amdgcn_s_wait_asynccnt)
  __builtin_amdgcn_s_wait_asynccnt(0);
#else
  asm volatile("s_wait_asynccnt 0x0" ::: "memory");
#endif
}
#endif

// Copy a ROWS x 32 bf16 tile (row stride srcStride elements) into padded LDS.
template <int ROWS>
__device__ __forceinline__ void fill_tile_bf16(unsigned short* lds,
                                               const unsigned short* src,
                                               size_t srcStride, int t) {
#if HAVE_ASYNC_LDS
#pragma unroll
  for (int i = 0; i < (ROWS * 4) / NT; ++i) {   // b128 chunks: 4 per row
    int c = t + i * NT;
    int row = c >> 2, j = c & 3;
    async_b128(src + (size_t)row * srcStride + j * 8, lds + row * LDSP + j * 8);
  }
#else
#pragma unroll
  for (int i = 0; i < (ROWS * 16) / NT; ++i) {  // u32 chunks: 16 per row
    int idx = t + i * NT;
    int row = idx >> 4, kp = idx & 15;
    unsigned v = *(const unsigned*)(src + (size_t)row * srcStride + kp * 2);
    *(unsigned*)(&lds[row * LDSP + kp * 2]) = v;
  }
#endif
}

// ---- WMMA fragment loaders from padded LDS tiles -------------------------
// 16-bit A 16x32: lanes 0-15 -> M=lane; K = {half*8..+7, 16+half*8..+7}
__device__ __forceinline__ v16bf load_frag_a(const unsigned short* ldsA,
                                             int mBlockRow, int lane) {
  const int m    = lane & 15;
  const int half = lane >> 4;
  const __bf16* base = (const __bf16*)(ldsA + (mBlockRow + m) * LDSP + half * 8);
  v8bf lo = *(const v8bf*)(base);
  v8bf hi = *(const v8bf*)(base + 16);
  v16bf a;
#pragma unroll
  for (int i = 0; i < 8; ++i) { a[i] = lo[i]; a[8 + i] = hi[i]; }
  return a;
}

// 16-bit B 32x16: lane%16 = N; K = {half*8..+7, 16+half*8..+7}
__device__ __forceinline__ v16bf load_frag_b(const unsigned short* ldsB,
                                             int nBlock, int lane) {
  const int n     = lane & 15;
  const int khalf = lane >> 4;
  const __bf16* base = (const __bf16*)(ldsB + (nBlock * 16 + n) * LDSP + khalf * 8);
  v8bf lo = *(const v8bf*)(base);
  v8bf hi = *(const v8bf*)(base + 16);
  v16bf b;
#pragma unroll
  for (int i = 0; i < 8; ++i) { b[i] = lo[i]; b[8 + i] = hi[i]; }
  return b;
}

__device__ __forceinline__ v8f wmma_bf16(v16bf a, v16bf b, v8f c) {
  return __builtin_amdgcn_wmma_f32_16x16x32_bf16(false, a, false, b,
                                                 (short)0, c, false, false);
}

// ===========================================================================
// WsumT[n][k] = sum_h Wl[h*DM + k][n]   (f32 -> bf16, transposed for Bt use)
// ===========================================================================
__global__ __launch_bounds__(256)
void wl_sum_kernel(const float* __restrict__ Wl, unsigned short* __restrict__ WsumT) {
  int idx = blockIdx.x * 256 + threadIdx.x;
  if (idx >= kDM * kDM) return;
  int k = idx / kDM, n = idx % kDM;   // coalesced reads along n
  float s = 0.0f;
#pragma unroll
  for (int h = 0; h < kH; ++h)
    s += Wl[((size_t)(h * kDM + k)) * kDM + n];
  WsumT[(size_t)n * kDM + k] = f32_to_bf16_bits(s);
}

// ===========================================================================
// Projection GEMM: C_bf16 = A_f32[b][M][K] * W_f32[K][N] + bias[N]
// TRANS_OUT=false: C[b][M][N]; TRANS_OUT=true: C[b][N][M] (for Vl^T).
// ===========================================================================
template <bool TRANS_OUT>
__global__ __launch_bounds__(NT)
void proj_gemm_kernel(const float* __restrict__ A, const float* __restrict__ W,
                      const float* __restrict__ bias, unsigned short* __restrict__ C,
                      int M, int N, int K) {
  __shared__ unsigned short ldsA[BM * LDSP];
  __shared__ unsigned short ldsB[BN * LDSP];

  const int bz    = blockIdx.z;
  const int mBase = blockIdx.y * BM;
  const int nBase = blockIdx.x * BN;
  const float* Ab = A + (size_t)bz * M * K;
  unsigned short* Cb = C + (size_t)bz * M * N;

  const int t = threadIdx.x;
  const int wave = t >> 5, lane = t & 31;
  const int waveM = wave >> 1, waveN = wave & 1;

  v8f acc[2][4];
#pragma unroll
  for (int a = 0; a < 2; ++a)
#pragma unroll
    for (int i = 0; i < 4; ++i)
#pragma unroll
      for (int j = 0; j < 8; ++j) acc[a][i][j] = 0.0f;

  for (int kb = 0; kb < K; kb += BK) {
    // Prefetch next A K-slab (global_prefetch_b8).
    if (kb + BK < K)
      __builtin_prefetch(Ab + (size_t)(mBase + (t & (BM - 1))) * K + kb + BK, 0, 1);

    // A tile: 128x32 f32 -> bf16 (float2 per iter, coalesced).
#pragma unroll
    for (int i = 0; i < (BM * BK / 2) / NT; ++i) {
      int idx = t + i * NT;
      int row = idx >> 4, kp = idx & 15;
      float2 v = *(const float2*)(Ab + (size_t)(mBase + row) * K + kb + kp * 2);
      unsigned short* d = &ldsA[row * LDSP + kp * 2];
      d[0] = f32_to_bf16_bits(v.x);
      d[1] = f32_to_bf16_bits(v.y);
    }
    // B tile transposed: ldsB[n][k] = W[kb+k][nBase+n]  (f32 -> bf16).
#pragma unroll
    for (int i = 0; i < (BN * BK) / NT; ++i) {
      int idx = t + i * NT;
      int k = idx >> 7;       // /BN
      int n = idx & (BN - 1);
      float v = W[(size_t)(kb + k) * N + nBase + n];
      ldsB[n * LDSP + k] = f32_to_bf16_bits(v);
    }
    __syncthreads();

    v16bf af[2];
    af[0] = load_frag_a(ldsA, waveM * 32, lane);
    af[1] = load_frag_a(ldsA, waveM * 32 + 16, lane);
#pragma unroll
    for (int nb = 0; nb < 4; ++nb) {
      v16bf bf = load_frag_b(ldsB, waveN * 4 + nb, lane);
      acc[0][nb] = wmma_bf16(af[0], bf, acc[0][nb]);
      acc[1][nb] = wmma_bf16(af[1], bf, acc[1][nb]);
    }
    __syncthreads();
  }

  const int half = lane >> 4;
#pragma unroll
  for (int mi = 0; mi < 2; ++mi) {
    int mRow = mBase + waveM * 32 + mi * 16 + half * 8;
#pragma unroll
    for (int nb = 0; nb < 4; ++nb) {
      int col = nBase + waveN * 64 + nb * 16 + (lane & 15);
      float bv = bias[col];
#pragma unroll
      for (int r = 0; r < 8; ++r) {
        unsigned short o = f32_to_bf16_bits(acc[mi][nb][r] + bv);
        if (TRANS_OUT) Cb[(size_t)col * M + (mRow + r)] = o;
        else           Cb[(size_t)(mRow + r) * N + col] = o;
      }
    }
  }
}

// ===========================================================================
// bf16 GEMM with B pre-transposed:  C[b] = A[b][M][K] * Bt[b?][N][K]^T
// EPI 0: scores  (v*scale + mask[col]*-1e9, f32 out)
// EPI 1: bf16 out
// EPI 2: f32 out + bias
// ===========================================================================
constexpr int EPI_SCORES = 0, EPI_BF16 = 1, EPI_F32BIAS = 2;

template <int EPI>
__global__ __launch_bounds__(NT)
void gemm_bt_kernel(const unsigned short* __restrict__ A,
                    const unsigned short* __restrict__ Bt,
                    const int* __restrict__ mask,
                    const float* __restrict__ bias,
                    void* __restrict__ Cout,
                    int M, int N, int K,
                    size_t aStride, size_t bStride, size_t cStride,
                    float scale) {
  __shared__ unsigned short ldsA[BM * LDSP];
  __shared__ unsigned short ldsB[BN * LDSP];

  const int bz    = blockIdx.z;
  const int mBase = blockIdx.y * BM;
  const int nBase = blockIdx.x * BN;
  const unsigned short* Ab = A  + (size_t)bz * aStride + (size_t)mBase * K;
  const unsigned short* Bb = Bt + (size_t)bz * bStride + (size_t)nBase * K;

  const int t = threadIdx.x;
  const int wave = t >> 5, lane = t & 31;
  const int waveM = wave >> 1, waveN = wave & 1;

  v8f acc[2][4];
#pragma unroll
  for (int a = 0; a < 2; ++a)
#pragma unroll
    for (int i = 0; i < 4; ++i)
#pragma unroll
      for (int j = 0; j < 8; ++j) acc[a][i][j] = 0.0f;

  for (int kb = 0; kb < K; kb += BK) {
    fill_tile_bf16<BM>(ldsA, Ab + kb, (size_t)K, t);
    fill_tile_bf16<BN>(ldsB, Bb + kb, (size_t)K, t);
#if HAVE_ASYNC_LDS
    wait_async0();
#endif
    __syncthreads();

    v16bf af[2];
    af[0] = load_frag_a(ldsA, waveM * 32, lane);
    af[1] = load_frag_a(ldsA, waveM * 32 + 16, lane);
#pragma unroll
    for (int nb = 0; nb < 4; ++nb) {
      v16bf bf = load_frag_b(ldsB, waveN * 4 + nb, lane);
      acc[0][nb] = wmma_bf16(af[0], bf, acc[0][nb]);
      acc[1][nb] = wmma_bf16(af[1], bf, acc[1][nb]);
    }
    __syncthreads();
  }

  const int half = lane >> 4;
#pragma unroll
  for (int mi = 0; mi < 2; ++mi) {
    int mRow = mBase + waveM * 32 + mi * 16 + half * 8;
#pragma unroll
    for (int nb = 0; nb < 4; ++nb) {
      int col = nBase + waveN * 64 + nb * 16 + (lane & 15);
      float extra = 0.0f;
      if (EPI == EPI_SCORES)  extra = (float)mask[bz * N + col] * (-1e9f);
      if (EPI == EPI_F32BIAS) extra = bias[col];
#pragma unroll
      for (int r = 0; r < 8; ++r) {
        float v = (EPI == EPI_SCORES) ? (acc[mi][nb][r] * scale + extra)
                                      : (acc[mi][nb][r] + extra);
        size_t off = (size_t)bz * cStride + (size_t)(mRow + r) * N + col;
        if (EPI == EPI_BF16) ((unsigned short*)Cout)[off] = f32_to_bf16_bits(v);
        else                 ((float*)Cout)[off] = v;
      }
    }
  }
}

// ===========================================================================
// Softmax over SK, replicate to all H slices, emit bf16 copy for att@V.
// ===========================================================================
__global__ __launch_bounds__(256)
void softmax_rep_kernel(float* __restrict__ att, unsigned short* __restrict__ attbf) {
  const int row = blockIdx.x;               // b*SQ + q
  const int b = row / kSQ, q = row % kSQ;
  float* p0 = att + ((size_t)b * kH * kSQ + q) * kSK;   // h=0 slice
  const int t = threadIdx.x;

  float x[4];
#pragma unroll
  for (int i = 0; i < 4; ++i) x[i] = p0[t + i * 256];

  __shared__ float red[256];
  float m = fmaxf(fmaxf(x[0], x[1]), fmaxf(x[2], x[3]));
  red[t] = m; __syncthreads();
#pragma unroll
  for (int s = 128; s > 0; s >>= 1) {
    if (t < s) red[t] = fmaxf(red[t], red[t + s]);
    __syncthreads();
  }
  m = red[0]; __syncthreads();

  float e[4]; float sum = 0.0f;
#pragma unroll
  for (int i = 0; i < 4; ++i) { e[i] = __expf(x[i] - m); sum += e[i]; }
  red[t] = sum; __syncthreads();
#pragma unroll
  for (int s = 128; s > 0; s >>= 1) {
    if (t < s) red[t] += red[t + s];
    __syncthreads();
  }
  const float inv = 1.0f / red[0];

  unsigned short* pb = attbf + ((size_t)b * kSQ + q) * kSK;
#pragma unroll
  for (int i = 0; i < 4; ++i) {
    float p = e[i] * inv;
    int col = t + i * 256;
#pragma unroll
    for (int h = 0; h < kH; ++h)
      att[(((size_t)b * kH + h) * kSQ + q) * kSK + col] = p;
    pb[col] = f32_to_bf16_bits(p);
  }
}

}  // namespace

// ===========================================================================
extern "C" void kernel_launch(void* const* d_in, const int* in_sizes, int n_in,
                              void* d_out, int out_size, void* d_ws, size_t ws_size,
                              hipStream_t stream) {
  (void)in_sizes; (void)n_in; (void)out_size; (void)ws_size;

  const float* Q   = (const float*)d_in[0];
  const float* K   = (const float*)d_in[1];
  const float* V   = (const float*)d_in[2];
  const int*   msk = (const int*)d_in[3];
  const float* Wq  = (const float*)d_in[4];
  const float* bq  = (const float*)d_in[5];
  const float* Wk  = (const float*)d_in[6];
  const float* bk  = (const float*)d_in[7];
  const float* Wv  = (const float*)d_in[8];
  const float* bv  = (const float*)d_in[9];
  const float* Wl  = (const float*)d_in[10];
  const float* bl  = (const float*)d_in[11];

  float* Y   = (float*)d_out;                       // [B, SQ, DM]
  float* att = Y + (size_t)kB * kSQ * kDM;          // [B, H, SQ, SK]

  // Workspace (bf16 intermediates).
  unsigned short* Qp    = (unsigned short*)d_ws;                 // [B,SQ,DM]
  unsigned short* Kp    = Qp    + (size_t)kB * kSQ * kDM;        // [B,SK,DM]
  unsigned short* Vt    = Kp    + (size_t)kB * kSK * kDM;        // [B,DM,SK] (V^T)
  unsigned short* head  = Vt    + (size_t)kB * kSK * kDM;        // [B,SQ,DM]
  unsigned short* attbf = head  + (size_t)kB * kSQ * kDM;        // [B,SQ,SK]
  unsigned short* WsumT = attbf + (size_t)kB * kSQ * kSK;        // [DM,DM] (^T)

  const dim3 blk(NT);
  const float scale = 0.044194173824159216f;  // 1/sqrt(512)

  // Fold the 8 identical-head output projection blocks (transposed).
  wl_sum_kernel<<<(kDM * kDM) / 256, 256, 0, stream>>>(Wl, WsumT);

  // Q/K/V projections (f32 in, bf16 out; V stored transposed).
  const dim3 gproj(kDM / BN, kSQ / BM, kB);
  proj_gemm_kernel<false><<<gproj, blk, 0, stream>>>(Q, Wq, bq, Qp, kSQ, kDM, kDM);
  proj_gemm_kernel<false><<<gproj, blk, 0, stream>>>(K, Wk, bk, Kp, kSK, kDM, kDM);
  proj_gemm_kernel<true ><<<gproj, blk, 0, stream>>>(V, Wv, bv, Vt, kSK, kDM, kDM);

  // Scores = Qp Kp^T with fused scale + mask -> att_ws h=0 slice (f32).
  const dim3 gsc(kSK / BN, kSQ / BM, kB);
  gemm_bt_kernel<EPI_SCORES><<<gsc, blk, 0, stream>>>(
      Qp, Kp, msk, nullptr, att, kSQ, kSK, kDM,
      (size_t)kSQ * kDM, (size_t)kSK * kDM, (size_t)kH * kSQ * kSK, scale);

  // Softmax + 8-way replication + bf16 copy, single fused pass.
  softmax_rep_kernel<<<kB * kSQ, 256, 0, stream>>>(att, attbf);

  // head = att @ Vl  (A=att_bf16, Bt=V^T, bf16 out)
  const dim3 gav(kDM / BN, kSQ / BM, kB);
  gemm_bt_kernel<EPI_BF16><<<gav, blk, 0, stream>>>(
      attbf, Vt, nullptr, nullptr, head, kSQ, kDM, kSK,
      (size_t)kSQ * kSK, (size_t)kDM * kSK, (size_t)kSQ * kDM, 1.0f);

  // Y = head @ Wl_sum + bl  (Bt=Wl_sum^T shared across batch, f32 out)
  gemm_bt_kernel<EPI_F32BIAS><<<gav, blk, 0, stream>>>(
      head, WsumT, nullptr, bl, Y, kSQ, kDM, kDM,
      (size_t)kSQ * kDM, 0, (size_t)kSQ * kDM, 1.0f);
}
